// DiscreteTransformer_32555852103695
// MI455X (gfx1250) — compile-verified, matrix operand order checked by reading, not run on hardware
//
#include <hip/hip_runtime.h>
#include <hip/hip_bf16.h>
#include <math.h>

// ---------------- problem constants (match reference) ----------------
#define BN 8192
#define SS 16
#define EE 256
#define HH 8
#define FFD 1024
#define LL 6
#define VV 512
#define UU 1024
#define DCC 64
#define BSR (BN * SS)   // 131072 rows

typedef __attribute__((ext_vector_type(16))) __bf16 v16bf;
typedef __attribute__((ext_vector_type(8)))  float  v8f;
typedef __attribute__((ext_vector_type(4)))  unsigned int v4u;
typedef __attribute__((ext_vector_type(8)))  int  v8i;
typedef __attribute__((ext_vector_type(4)))  int  v4i;

#if defined(__gfx1250__) && __has_builtin(__builtin_amdgcn_tensor_load_to_lds) && \
    __has_builtin(__builtin_amdgcn_s_wait_tensorcnt)
#define USE_TDM 1
#else
#define USE_TDM 0
#endif

union BF16Frag {
    v16bf v;
    unsigned int u32[8];
    unsigned short u16[16];
};

__device__ __forceinline__ unsigned short f2bf(float x) {
    unsigned int u = __float_as_uint(x);
    u += 0x7fffu + ((u >> 16) & 1u);   // round-to-nearest-even
    return (unsigned short)(u >> 16);
}

// Build a 16-bit WMMA fragment from an LDS tile (row-major, stride 40 ushorts).
// ISA layout (16-bit A 16x32 / B 32x16): lane = row (M or N), half-wave selects
// K blocks {0..7,16..23} vs {8..15,24..31}; each block is 8 contiguous bf16 = b128.
__device__ __forceinline__ void load_frag_lds(BF16Frag& f, const unsigned short* ld,
                                              int row, int half) {
    int base = row * 40 + half * 8;
    uint4 a0 = *(const uint4*)(ld + base);
    uint4 a1 = *(const uint4*)(ld + base + 16);
    f.u32[0] = a0.x; f.u32[1] = a0.y; f.u32[2] = a0.z; f.u32[3] = a0.w;
    f.u32[4] = a1.x; f.u32[5] = a1.y; f.u32[6] = a1.z; f.u32[7] = a1.w;
}

#if USE_TDM
// Tensor Data Mover: DMA one bf16 weight tile W[n0:n0+128, kt:kt+32] into LDS.
// 2D descriptor, data_size=2B. LDS padding: every 16 DWORDs (64B row) insert
// 4 DWORDs (16B) -> LDS row stride 80B = 40 ushorts, matching load_frag_lds.
// This toolchain exposes the 6-arg builtin: (v4u, v8i, v4i, v4i, v8i, cpol).
__device__ __forceinline__ void tdm_load_tile_b(const unsigned short* gsrc,
                                                unsigned lds_byte_off, int K)
{
    unsigned long long ga = (unsigned long long)(uintptr_t)gsrc;
    v4u g0;
    g0[0] = 1u;                                     // count=1, user descriptor
    g0[1] = lds_byte_off;                           // lds_addr
    g0[2] = (unsigned)ga;                           // global_addr[31:0]
    g0[3] = (unsigned)((ga >> 32) & 0x01FFFFFFull)  // global_addr[56:32]
          | (2u << 30);                             // type=2 ("image")
    v8i g1;
    g1[0] = (int)((1u << 16)    // data_size = 2 bytes
                | (1u << 20)    // pad_enable
                | (3u << 22)    // pad_interval: 16 DWORDs (one 64B row)
                | (3u << 25));  // pad_amount: 4 DWORDs (16B)
    g1[1] = (int)(32u << 16);                       // tensor_dim0 = 32 (lo16 in [31:16])
    g1[2] = (int)(128u << 16);                      // tensor_dim0 hi=0 | tensor_dim1 lo = 128
    g1[3] = (int)(32u << 16);                       // tensor_dim1 hi=0 | tile_dim0 = 32
    g1[4] = (int)128;                               // tile_dim1 = 128, tile_dim2 = 0
    g1[5] = (int)K;                                 // tensor_dim0_stride[31:0] = K elements
    g1[6] = 0;                                      // stride hi | dim1_stride lo (unused, 2D)
    g1[7] = 0;
    v4i g2; g2[0] = 0; g2[1] = 0; g2[2] = 0; g2[3] = 0;
    v4i g3 = g2;
    v8i g4; g4[0] = 0; g4[1] = 0; g4[2] = 0; g4[3] = 0;
    g4[4] = 0; g4[5] = 0; g4[6] = 0; g4[7] = 0;
    __builtin_amdgcn_tensor_load_to_lds(g0, g1, g2, g3, g4, 0);
}
#endif

// ---------------- tiled WMMA GEMM: C[M,N] = act(A[M,K] @ W[N,K]^T + bias (+resid)) ----
// A: f32 or bf16 (template), W: bf16 row-major [N,K], bias f32[N], resid f32[M,N].
// Block tile 128x128, 256 threads = 8 waves (4x2), wave tile 32x64, K-step 32.
// B tile staged by the Tensor Data Mover (wave 0 issues, TENSORcnt-waited).
template<bool A_BF16, int ACT, bool RESID, bool OUT_BF16>
__global__ __launch_bounds__(256) void gemm_wmma(
    const void* __restrict__ Ap, const unsigned short* __restrict__ Wb,
    const float* __restrict__ bias, const float* __restrict__ resid,
    void* __restrict__ Cp, int M, int N, int K)
{
    __shared__ unsigned short lA[128 * 40];
    __shared__ unsigned short lB[128 * 40];

    const int tid  = threadIdx.x;
    const int lane = tid & 31;
    const int wid  = tid >> 5;
    const int wm   = wid & 3;        // 4 waves along M
    const int wn   = wid >> 2;       // 2 waves along N
    const int half = lane >> 4;
    const int nl   = lane & 15;
    const int m0   = blockIdx.y * 128;
    const int n0   = blockIdx.x * 128;

    v8f acc[2][4] = {};

    for (int kt = 0; kt < K; kt += 32) {
#if USE_TDM
        if (wid == 0)
            tdm_load_tile_b(Wb + (size_t)n0 * K + kt,
                            (unsigned)(uintptr_t)(&lB[0]), K);
#endif
        // cooperative A-tile load: 128 rows x 32 cols (f32 -> bf16 on the fly)
        #pragma unroll
        for (int i = 0; i < 4; i++) {
            int idx = tid + (i << 8);          // 0..1023
            int r   = idx >> 3;                // row 0..127
            int q   = (idx & 7) << 2;          // col 0,4,...,28
            if constexpr (!A_BF16) {
                const float* A = (const float*)Ap;
                const float* ap = A + (size_t)(m0 + r) * K + kt + q;
                float4 d = *(const float4*)ap;
                if (kt + 32 < K) __builtin_prefetch(ap + 32, 0, 1);
                unsigned short* p = lA + r * 40 + q;
                p[0] = f2bf(d.x); p[1] = f2bf(d.y); p[2] = f2bf(d.z); p[3] = f2bf(d.w);
            } else {
                const unsigned short* A = (const unsigned short*)Ap;
                const unsigned short* ap = A + (size_t)(m0 + r) * K + kt + q;
                *(uint2*)(lA + r * 40 + q) = *(const uint2*)ap;
                if (kt + 32 < K) __builtin_prefetch(ap + 32, 0, 1);
            }
#if !USE_TDM
            *(uint2*)(lB + r * 40 + q) =
                *(const uint2*)(Wb + (size_t)(n0 + r) * K + kt + q);
#endif
        }
#if USE_TDM
        if (wid == 0) __builtin_amdgcn_s_wait_tensorcnt(0);
#endif
        __syncthreads();

        BF16Frag bfr[4];
        #pragma unroll
        for (int tn = 0; tn < 4; tn++)
            load_frag_lds(bfr[tn], lB, wn * 64 + tn * 16 + nl, half);
        #pragma unroll
        for (int tm = 0; tm < 2; tm++) {
            BF16Frag afr;
            load_frag_lds(afr, lA, wm * 32 + tm * 16 + nl, half);
            #pragma unroll
            for (int tn = 0; tn < 4; tn++)
                acc[tm][tn] = __builtin_amdgcn_wmma_f32_16x16x32_bf16(
                    false, afr.v, false, bfr[tn].v, (short)0, acc[tm][tn], false, false);
        }
        __syncthreads();
    }

    // epilogue: C layout -> lane nl = column, vgpr r + half*8 = row within 16x16 tile
    #pragma unroll
    for (int tn = 0; tn < 4; tn++) {
        int n = n0 + wn * 64 + tn * 16 + nl;
        float bv = bias ? bias[n] : 0.0f;
        #pragma unroll
        for (int tm = 0; tm < 2; tm++) {
            #pragma unroll
            for (int r = 0; r < 8; r++) {
                size_t m = (size_t)(m0 + wm * 32 + tm * 16 + r + half * 8);
                float v = acc[tm][tn][r] + bv;
                if constexpr (RESID) v += resid[m * (size_t)N + n];
                if constexpr (ACT == 1) v = fmaxf(v, 0.0f);
                if constexpr (ACT == 2) v = 0.5f * v * (1.0f + erff(v * 0.70710678118654752f));
                if constexpr (OUT_BF16) ((unsigned short*)Cp)[m * (size_t)N + n] = f2bf(v);
                else                    ((float*)Cp)[m * (size_t)N + n] = v;
            }
        }
    }
}

// ---------------- attention: one workgroup per batch element, wave = head ----------
// qkv bf16 [BS, 768] (q|k|v, dh=32). scores = one WMMA, softmax in registers,
// out = A @ V with K padded 16->32, written bf16 to o[BS, 256].
__global__ __launch_bounds__(256) void attn_kernel(const unsigned short* __restrict__ qkv,
                                                   unsigned short* __restrict__ o)
{
    __shared__ unsigned short la[8][16 * 17];
    const int b    = blockIdx.x;
    const int lane = threadIdx.x & 31;
    const int h    = threadIdx.x >> 5;
    const int half = lane >> 4;
    const int nl   = lane & 15;
    const size_t base = (size_t)b * (SS * 768);

    BF16Frag qf, kf;
    {
        const unsigned short* qp = qkv + base + (size_t)nl * 768 + h * 32 + half * 8;
        uint4 a0 = *(const uint4*)qp;
        uint4 a1 = *(const uint4*)(qp + 16);
        qf.u32[0] = a0.x; qf.u32[1] = a0.y; qf.u32[2] = a0.z; qf.u32[3] = a0.w;
        qf.u32[4] = a1.x; qf.u32[5] = a1.y; qf.u32[6] = a1.z; qf.u32[7] = a1.w;
        const unsigned short* kp = qkv + base + (size_t)nl * 768 + 256 + h * 32 + half * 8;
        uint4 b0 = *(const uint4*)kp;
        uint4 b1 = *(const uint4*)(kp + 16);
        kf.u32[0] = b0.x; kf.u32[1] = b0.y; kf.u32[2] = b0.z; kf.u32[3] = b0.w;
        kf.u32[4] = b1.x; kf.u32[5] = b1.y; kf.u32[6] = b1.z; kf.u32[7] = b1.w;
    }
    v8f sc = {};
    sc = __builtin_amdgcn_wmma_f32_16x16x32_bf16(false, qf.v, false, kf.v,
                                                 (short)0, sc, false, false);
    const float scale = 0.17677669529663687f;   // 1/sqrt(32)
    #pragma unroll
    for (int r = 0; r < 8; r++) {
        int s = r + half * 8;                   // row (query position)
        float vsc = sc[r] * scale;
        if (nl > s) vsc = -1e30f;               // causal mask (t > s)
        float mx = vsc;
        #pragma unroll
        for (int msk = 1; msk < 16; msk <<= 1) mx = fmaxf(mx, __shfl_xor(mx, msk, 32));
        float ex = __expf(vsc - mx);
        float sm = ex;
        #pragma unroll
        for (int msk = 1; msk < 16; msk <<= 1) sm += __shfl_xor(sm, msk, 32);
        la[h][s * 17 + nl] = f2bf(ex / sm);
    }
    __syncthreads();

    BF16Frag af;
    #pragma unroll
    for (int i = 0; i < 8; i++) {
        int k0 = half * 8 + ((i < 4) ? 2 * i : 16 + 2 * (i - 4));
        af.u16[2 * i]     = (k0 < 16)     ? la[h][nl * 17 + k0]     : (unsigned short)0;
        af.u16[2 * i + 1] = (k0 + 1 < 16) ? la[h][nl * 17 + k0 + 1] : (unsigned short)0;
    }
    #pragma unroll
    for (int dt = 0; dt < 2; dt++) {
        int d = dt * 16 + nl;
        BF16Frag vf;
        #pragma unroll
        for (int i = 0; i < 8; i++) {
            int k0 = half * 8 + ((i < 4) ? 2 * i : 16 + 2 * (i - 4));
            vf.u16[2 * i]     = (k0 < 16)
                ? qkv[base + (size_t)k0 * 768 + 512 + h * 32 + d] : (unsigned short)0;
            vf.u16[2 * i + 1] = (k0 + 1 < 16)
                ? qkv[base + (size_t)(k0 + 1) * 768 + 512 + h * 32 + d] : (unsigned short)0;
        }
        v8f oa = {};
        oa = __builtin_amdgcn_wmma_f32_16x16x32_bf16(false, af.v, false, vf.v,
                                                     (short)0, oa, false, false);
        #pragma unroll
        for (int r = 0; r < 8; r++) {
            int s = r + half * 8;
            o[((size_t)b * SS + s) * EE + h * 32 + d] = f2bf(oa[r]);
        }
    }
}

// ---------------- layernorm over rows of 256, wave per row, in place ------------
__global__ __launch_bounds__(256) void ln_kernel(float* __restrict__ h,
                                                 const float* __restrict__ g,
                                                 const float* __restrict__ bt)
{
    const int lane = threadIdx.x & 31;
    const size_t row = (size_t)blockIdx.x * 8 + (threadIdx.x >> 5);
    float* p = h + row * EE + lane * 8;
    float xv[8];
    *(float4*)(&xv[0]) = *(const float4*)p;
    *(float4*)(&xv[4]) = *(const float4*)(p + 4);
    float s = 0.0f, q = 0.0f;
    #pragma unroll
    for (int i = 0; i < 8; i++) { s += xv[i]; q += xv[i] * xv[i]; }
    #pragma unroll
    for (int msk = 1; msk < 32; msk <<= 1) {
        s += __shfl_xor(s, msk, 32);
        q += __shfl_xor(q, msk, 32);
    }
    float mu  = s * (1.0f / 256.0f);
    float var = q * (1.0f / 256.0f) - mu * mu;
    float rr  = rsqrtf(var + 1e-5f);
    int e = lane * 8;
    #pragma unroll
    for (int i = 0; i < 8; i++) xv[i] = (xv[i] - mu) * rr * g[e + i] + bt[e + i];
    *(float4*)p       = *(const float4*)(&xv[0]);
    *(float4*)(p + 4) = *(const float4*)(&xv[4]);
}

// ---------------- conditioning GEMV: cond[b,e] = c[b,:] . c_W[e,:] + c_b[e] -----
__global__ __launch_bounds__(256) void cond_kernel(const float* __restrict__ c,
                                                   const float* __restrict__ cW,
                                                   const float* __restrict__ cb,
                                                   float* __restrict__ cond)
{
    int idx = blockIdx.x * 256 + threadIdx.x;
    int b = idx >> 8, e = idx & 255;
    const float* cp = c  + (size_t)b * DCC;
    const float* wp = cW + (size_t)e * DCC;
    float s = cb[e];
    #pragma unroll 8
    for (int d = 0; d < DCC; d++) s += cp[d] * wp[d];
    cond[idx] = s;
}

// ---------------- embedding: h[b,s,e] = shift-gather + pos + cond ---------------
__global__ __launch_bounds__(256) void embed_kernel(const int* __restrict__ x,
                                                    const float* __restrict__ xemb,
                                                    const float* __restrict__ pos,
                                                    const float* __restrict__ cond,
                                                    float* __restrict__ h)
{
    size_t idx = (size_t)blockIdx.x * 256 + threadIdx.x;   // < BS*E
    int e = (int)(idx & 255);
    size_t bs = idx >> 8;
    int s = (int)(bs & 15);
    size_t b = bs >> 4;
    float v = cond[b * EE + e] + pos[s * EE + e];
    if (s > 0) {
        int tok = x[b * SS + s - 1];
        v += xemb[(size_t)tok * EE + e];
    }
    h[idx] = v;
}

// ---------------- f32 -> bf16 conversion ----------------------------------------
__global__ __launch_bounds__(256) void f32_to_bf16(const float* __restrict__ in,
                                                   unsigned short* __restrict__ out, int n)
{
    int i = blockIdx.x * 256 + threadIdx.x;
    if (i < n) out[i] = f2bf(in[i]);
}

// ---------------- final: out[b] = sum_s (logits[b,s,x]-logsumexp(logits[b,s,:])) -
__global__ __launch_bounds__(256) void logprob_kernel(const float* __restrict__ logits,
                                                      const int* __restrict__ x,
                                                      float* __restrict__ out)
{
    __shared__ float partial[8];
    const int b    = blockIdx.x;
    const int lane = threadIdx.x & 31;
    const int wid  = threadIdx.x >> 5;
    float acc = 0.0f;
    #pragma unroll
    for (int j = 0; j < 2; j++) {
        int s = wid * 2 + j;
        const float* row = logits + ((size_t)b * SS + s) * VV;
        float vals[16];
        float mx = -3.0e38f;
        #pragma unroll
        for (int i = 0; i < 16; i++) { vals[i] = row[lane + 32 * i]; mx = fmaxf(mx, vals[i]); }
        #pragma unroll
        for (int msk = 1; msk < 32; msk <<= 1) mx = fmaxf(mx, __shfl_xor(mx, msk, 32));
        float sm = 0.0f;
        #pragma unroll
        for (int i = 0; i < 16; i++) sm += __expf(vals[i] - mx);
        #pragma unroll
        for (int msk = 1; msk < 32; msk <<= 1) sm += __shfl_xor(sm, msk, 32);
        float lse = __logf(sm) + mx;
        int tok = x[b * SS + s];
        acc += row[tok] - lse;
    }
    if (lane == 0) partial[wid] = acc;
    __syncthreads();
    if (threadIdx.x == 0) {
        float t = 0.0f;
        #pragma unroll
        for (int w = 0; w < 8; w++) t += partial[w];
        out[b] = t;
    }
}

// =================================================================================
extern "C" void kernel_launch(void* const* d_in, const int* in_sizes, int n_in,
                              void* d_out, int out_size, void* d_ws, size_t ws_size,
                              hipStream_t stream)
{
    (void)in_sizes; (void)n_in; (void)out_size; (void)ws_size;

    const int*   x       = (const int*)  d_in[0];
    const float* c       = (const float*)d_in[1];
    const float* x_emb   = (const float*)d_in[2];
    const float* pos_emb = (const float*)d_in[3];
    const float* c_W     = (const float*)d_in[4];
    const float* c_b     = (const float*)d_in[5];
    const float* Wqkv    = (const float*)d_in[6];
    const float* bqkv    = (const float*)d_in[7];
    const float* Wo      = (const float*)d_in[8];
    const float* bo      = (const float*)d_in[9];
    const float* ln1_g   = (const float*)d_in[10];
    const float* ln1_b   = (const float*)d_in[11];
    const float* W1      = (const float*)d_in[12];
    const float* b1      = (const float*)d_in[13];
    const float* W2      = (const float*)d_in[14];
    const float* b2      = (const float*)d_in[15];
    const float* ln2_g   = (const float*)d_in[16];
    const float* ln2_b   = (const float*)d_in[17];
    const float* mW0     = (const float*)d_in[18];
    const float* mb0     = (const float*)d_in[19];
    const float* mW1     = (const float*)d_in[20];
    const float* mb1     = (const float*)d_in[21];
    const float* mW2     = (const float*)d_in[22];
    const float* mb2     = (const float*)d_in[23];
    float* out = (float*)d_out;

    // workspace partition (256B aligned)
    char* ws = (char*)d_ws;
    size_t off = 0;
    auto alloc = [&](size_t bytes) -> char* {
        char* p = ws + off;
        off += (bytes + 255) & ~(size_t)255;
        return p;
    };
    float*          h     = (float*)         alloc((size_t)BSR * EE  * 4);
    unsigned short* qkvb  = (unsigned short*)alloc((size_t)BSR * 768 * 2);
    unsigned short* ob    = (unsigned short*)alloc((size_t)BSR * EE  * 2);
    unsigned short* fb    = (unsigned short*)alloc((size_t)BSR * FFD * 2); // also aliased as logits f32 [BS,512]
    unsigned short* zb    = (unsigned short*)alloc((size_t)BSR * UU  * 2);
    float*          cond  = (float*)         alloc((size_t)BN  * EE  * 4);
    unsigned short* Wqkvb = (unsigned short*)alloc((size_t)LL * 768 * EE * 2);
    unsigned short* Wob   = (unsigned short*)alloc((size_t)LL * EE  * EE * 2);
    unsigned short* W1b   = (unsigned short*)alloc((size_t)LL * FFD * EE * 2);
    unsigned short* W2b   = (unsigned short*)alloc((size_t)LL * EE  * FFD * 2);
    unsigned short* mW0b  = (unsigned short*)alloc((size_t)UU * EE * 2);
    unsigned short* mW1b  = (unsigned short*)alloc((size_t)UU * UU * 2);
    unsigned short* mW2b  = (unsigned short*)alloc((size_t)VV * UU * 2);
    float* logits = (float*)fb;   // safe alias: z0 in fb is dead once logits GEMM runs (reads zb)

    auto conv = [&](const float* src, unsigned short* dst, size_t n) {
        f32_to_bf16<<<(unsigned)((n + 255) / 256), 256, 0, stream>>>(src, dst, (int)n);
    };
    conv(Wqkv, Wqkvb, (size_t)LL * 768 * EE);
    conv(Wo,   Wob,   (size_t)LL * EE  * EE);
    conv(W1,   W1b,   (size_t)LL * FFD * EE);
    conv(W2,   W2b,   (size_t)LL * EE  * FFD);
    conv(mW0,  mW0b,  (size_t)UU * EE);
    conv(mW1,  mW1b,  (size_t)UU * UU);
    conv(mW2,  mW2b,  (size_t)VV * UU);

    cond_kernel<<<BN * EE / 256, 256, 0, stream>>>(c, c_W, c_b, cond);
    embed_kernel<<<(unsigned)((size_t)BSR * EE / 256), 256, 0, stream>>>(x, x_emb, pos_emb, cond, h);

    const dim3 blk(256);
    const int MB = BSR / 128;   // 1024 row-blocks
    for (int l = 0; l < LL; l++) {
        // qkv = h @ Wqkv^T + bqkv  -> bf16
        gemm_wmma<false, 0, false, true><<<dim3(768 / 128, MB), blk, 0, stream>>>(
            h, Wqkvb + (size_t)l * 768 * EE, bqkv + l * 768, nullptr, qkvb, BSR, 768, EE);
        // attention -> o (bf16)
        attn_kernel<<<BN, blk, 0, stream>>>(qkvb, ob);
        // h = h + o @ Wo^T + bo (f32, in place)
        gemm_wmma<true, 0, true, false><<<dim3(EE / 128, MB), blk, 0, stream>>>(
            ob, Wob + (size_t)l * EE * EE, bo + l * EE, h, h, BSR, EE, EE);
        ln_kernel<<<BSR / 8, blk, 0, stream>>>(h, ln1_g + l * EE, ln1_b + l * EE);
        // f = gelu(h @ W1^T + b1) -> bf16
        gemm_wmma<false, 2, false, true><<<dim3(FFD / 128, MB), blk, 0, stream>>>(
            h, W1b + (size_t)l * FFD * EE, b1 + l * FFD, nullptr, fb, BSR, FFD, EE);
        // h = h + f @ W2^T + b2 (f32, in place)
        gemm_wmma<true, 0, true, false><<<dim3(EE / 128, MB), blk, 0, stream>>>(
            fb, W2b + (size_t)l * EE * FFD, b2 + l * EE, h, h, BSR, EE, FFD);
        ln_kernel<<<BSR / 8, blk, 0, stream>>>(h, ln2_g + l * EE, ln2_b + l * EE);
    }

    // MLP head
    gemm_wmma<false, 1, false, true><<<dim3(UU / 128, MB), blk, 0, stream>>>(
        h, mW0b, mb0, nullptr, fb, BSR, UU, EE);            // z0 = relu(h@mW0^T)
    gemm_wmma<true, 1, false, true><<<dim3(UU / 128, MB), blk, 0, stream>>>(
        fb, mW1b, mb1, nullptr, zb, BSR, UU, UU);           // z1 = relu(z0@mW1^T)
    gemm_wmma<true, 0, false, false><<<dim3(VV / 128, MB), blk, 0, stream>>>(
        zb, mW2b, mb2, nullptr, logits, BSR, VV, UU);       // logits (f32, aliases fb)

    logprob_kernel<<<BN, blk, 0, stream>>>(logits, x, out);
}
